// DualStoSelfAttention_24747601560232
// MI455X (gfx1250) — matrix operand order repaired and checked by reading, not compiled, not run on hardware
//
#include <hip/hip_runtime.h>
#include <stdint.h>

// ---------------- problem constants ----------------
#define Bb 8
#define Ss 1024
#define Hh 768
#define NHh 12
#define HDd 64
#define Mm (Bb * Ss)      // 8192 rows of token matrix
#define NQKV (3 * Hh)     // 2304

typedef __attribute__((ext_vector_type(16))) __bf16 v16bf;
typedef __attribute__((ext_vector_type(8)))  __bf16 v8bf;
typedef __attribute__((ext_vector_type(8)))  float  v8f;

// ---------------- helpers ----------------
__device__ inline v8f zero8() {
    v8f z;
#pragma unroll
    for (int i = 0; i < 8; ++i) z[i] = 0.0f;
    return z;
}

// deterministic hash -> uniform(0,1) -> gumbel
__device__ inline float gumbel_hash(unsigned seed, unsigned idx) {
    unsigned x = idx * 0x9E3779B1u + seed;
    x ^= x >> 16; x *= 0x7FEB352Du;
    x ^= x >> 15; x *= 0x846CA68Bu;
    x ^= x >> 16;
    float u = (float)(x >> 8) * (1.0f / 16777216.0f) + 1e-7f;   // (0,1)
    return -__logf(-__logf(u));
}

// async DMA: global (16B) -> LDS, no VGPR staging; tracked by ASYNCcnt
__device__ inline void async_g2l_b128(const void* gaddr, void* lds) {
    unsigned long long ga = (unsigned long long)(uintptr_t)gaddr;
    unsigned la = (unsigned)(uintptr_t)lds;              // LDS byte offset
    asm volatile("global_load_async_to_lds_b128 %0, %1, off"
                 :: "v"(la), "v"(ga) : "memory");
}
__device__ inline void wait_asynccnt0() {
    asm volatile("s_wait_asynccnt 0x0" ::: "memory");
}

// ---------------- conversion / packing ----------------
__global__ void f32_to_bf16_kernel(const float* __restrict__ src,
                                   __bf16* __restrict__ dst, int n) {
    for (int i = blockIdx.x * blockDim.x + threadIdx.x; i < n;
         i += gridDim.x * blockDim.x)
        dst[i] = (__bf16)src[i];
}

// Wcat[k][n] (row-major K=768 x N=2304) = [Wq | Wk | Wv], bcat = [bq|bk|bv]
__global__ void pack_wcat_kernel(const float* __restrict__ Wq,
                                 const float* __restrict__ Wk,
                                 const float* __restrict__ Wv,
                                 const float* __restrict__ bq,
                                 const float* __restrict__ bk,
                                 const float* __restrict__ bv,
                                 __bf16* __restrict__ Wcat,
                                 float* __restrict__ bcat) {
    int total = Hh * NQKV;
    for (int i = blockIdx.x * blockDim.x + threadIdx.x; i < total;
         i += gridDim.x * blockDim.x) {
        int k = i / NQKV, n = i % NQKV;
        float v;
        if (n < Hh)            v = Wq[k * Hh + n];
        else if (n < 2 * Hh)   v = Wk[k * Hh + (n - Hh)];
        else                   v = Wv[k * Hh + (n - 2 * Hh)];
        Wcat[i] = (__bf16)v;
        if (k == 0) {
            bcat[n] = (n < Hh) ? bq[n] : (n < 2 * Hh) ? bk[n - Hh] : bv[n - 2 * Hh];
        }
    }
}

// ---------------- WMMA GEMM: C[M,N] = A[M,K]*B[K,N] + bias ----------------
// A,B bf16 row-major; C fp32.
// Block: 256 thr (8 waves). Block tile 128(M) x 128(N), K-step 32.
// Wave (wy = wid&3, wx = wid>>2) owns 32(M) x 64(N): 2 A-frags x 4 B-frags = 8 WMMA.
// Double-buffered LDS; A tile via async global->LDS DMA (no VGPR staging),
// B tile via 2 int4 regs + transpose scatter. One barrier per K-step.
#define LDS_STRIDE 40   // halves; 80B rows -> 16B-aligned pieces, conflict-free b128

__global__ __launch_bounds__(256)
void gemm_bf16_wmma_kernel(const __bf16* __restrict__ A,
                           const __bf16* __restrict__ B,
                           const float* __restrict__ bias,
                           float* __restrict__ C,
                           int M, int N, int K, int lda, int ldb, int ldc) {
    __shared__ __bf16 lA[2][128 * LDS_STRIDE];
    __shared__ __bf16 lBT[2][128 * LDS_STRIDE];   // transposed: [n][k]

    const int tid  = threadIdx.x;
    const int wid  = tid >> 5;         // 0..7
    const int lane = tid & 31;
    const int g    = lane >> 4;        // lane half
    const int l16  = lane & 15;
    const int wy   = wid & 3;          // M sub-block (4 x 32 = 128)
    const int wx   = wid >> 2;         // N sub-block (2 x 64 = 128)

    const int mBlock = blockIdx.y * 128;
    const int nBlock = blockIdx.x * 128;

    v8f acc[2][4];
#pragma unroll
    for (int s = 0; s < 2; ++s)
#pragma unroll
        for (int j = 0; j < 4; ++j) acc[s][j] = zero8();

    const int T = K >> 5;              // K-tiles of 32

    // ---- A tile 128x32: 512 x 16B async DMA transfers per block ----
    const int arow = tid >> 2;                 // thread's A row (two chunks)
    const int aco0 = (tid & 3) * 8;            // 0,8,16,24 (chunk0)
    auto asyncA = [&](int k0, int buf) {
        async_g2l_b128(&A[(size_t)(mBlock + arow) * lda + k0 + aco0],
                       &lA[buf][arow * LDS_STRIDE + aco0]);
        int row1 = (tid + 256) >> 2;
        int co1  = ((tid + 256) & 3) * 8;
        async_g2l_b128(&A[(size_t)(mBlock + row1) * lda + k0 + co1],
                       &lA[buf][row1 * LDS_STRIDE + co1]);
    };

    // ---- B tile 32x128: 2 int4 regs per thread, transposed into LDS ----
    const int bk0 = tid >> 4;                  // 0..15
    const int bnb = (tid & 15) * 8;            // 0..120
    int4 br0, br1;
    auto loadB = [&](int k0) {
        br0 = *(const int4*)&B[(size_t)(k0 + bk0) * ldb + nBlock + bnb];
        br1 = *(const int4*)&B[(size_t)(k0 + bk0 + 16) * ldb + nBlock + bnb];
    };
    auto storeB = [&](int buf) {
        const __bf16* p0 = (const __bf16*)&br0;
        const __bf16* p1 = (const __bf16*)&br1;
#pragma unroll
        for (int j = 0; j < 8; ++j) {
            lBT[buf][(bnb + j) * LDS_STRIDE + bk0]      = p0[j];
            lBT[buf][(bnb + j) * LDS_STRIDE + bk0 + 16] = p1[j];
        }
    };

    // ---- prologue: tile 0 into buf0; B tile 1 into regs ----
    loadB(0);
    asyncA(0, 0);
    storeB(0);
    if (T > 1) loadB(32);
    wait_asynccnt0();
    __syncthreads();

    for (int it = 0; it < T; ++it) {
        const int buf = it & 1;
        // stage tile it+1 into buf^1 while computing tile it (DMA overlaps WMMA)
        if (it + 1 < T) {
            asyncA((it + 1) << 5, buf ^ 1);
            storeB(buf ^ 1);                       // regs hold tile it+1
            if (it + 2 < T) loadB((it + 2) << 5);  // prefetch B tile it+2
        }

        // ---- fragments + 8 WMMA ----
        v16bf af[2];
#pragma unroll
        for (int s = 0; s < 2; ++s) {
            int mrow = wy * 32 + s * 16 + l16;
            v8bf a0 = *(const v8bf*)&lA[buf][mrow * LDS_STRIDE + 8 * g];
            v8bf a1 = *(const v8bf*)&lA[buf][mrow * LDS_STRIDE + 16 + 8 * g];
            af[s] = __builtin_shufflevector(a0, a1, 0, 1, 2, 3, 4, 5, 6, 7,
                                            8, 9, 10, 11, 12, 13, 14, 15);
        }
#pragma unroll
        for (int j = 0; j < 4; ++j) {
            int nloc = wx * 64 + j * 16 + l16;
            v8bf b0 = *(const v8bf*)&lBT[buf][nloc * LDS_STRIDE + 8 * g];
            v8bf b1 = *(const v8bf*)&lBT[buf][nloc * LDS_STRIDE + 16 + 8 * g];
            v16bf bfr = __builtin_shufflevector(b0, b1, 0, 1, 2, 3, 4, 5, 6, 7,
                                                8, 9, 10, 11, 12, 13, 14, 15);
#pragma unroll
            for (int s = 0; s < 2; ++s)
                acc[s][j] = __builtin_amdgcn_wmma_f32_16x16x32_bf16(
                    false, af[s], false, bfr, (short)0, acc[s][j], false, false);
        }

        wait_asynccnt0();      // tile it+1 DMA complete (no-op on last iter)
        __syncthreads();
    }

    // ---- epilogue: C layout VGPR r -> M=r (lanes 0-15) / M=8+r (lanes 16-31) ----
#pragma unroll
    for (int s = 0; s < 2; ++s) {
        const int mrow0 = mBlock + wy * 32 + s * 16 + g * 8;
#pragma unroll
        for (int j = 0; j < 4; ++j) {
            int n = nBlock + wx * 64 + j * 16 + l16;
            float bv = bias ? bias[n] : 0.0f;
#pragma unroll
            for (int r = 0; r < 8; ++r)
                C[(size_t)(mrow0 + r) * ldc + n] = acc[s][j][r] + bv;
        }
    }
}

// ---------------- per-(token,head): key class + q-centroid dots ----------------
// Cqkv row m: cols [0,768)=Q, [768,1536)=K, [1536,2304)=V
__global__ void quant_qc_kernel(const float* __restrict__ Cqkv,
                                const float* __restrict__ centroid,  // [64][2]
                                float* __restrict__ qc,              // [M*NH][2]
                                uint8_t* __restrict__ kclass) {      // [M*NH]
    int item = blockIdx.x * blockDim.x + threadIdx.x;                // m*12+h
    if (item >= Mm * NHh) return;
    int m = item / NHh, h = item % NHh;
    const float* Qv = Cqkv + (size_t)m * NQKV + h * HDd;
    const float* Kv = Qv + Hh;
    float l0 = 0.f, l1 = 0.f, q0 = 0.f, q1 = 0.f;
#pragma unroll 8
    for (int d = 0; d < HDd; ++d) {
        float c0 = centroid[d * 2], c1 = centroid[d * 2 + 1];
        float kd = Kv[d], qd = Qv[d];
        l0 += kd * c0; l1 += kd * c1;
        q0 += qd * c0; q1 += qd * c1;
    }
    // hard gumbel top-1 over 2 centroids: argmax(logit + g) (tau>0 irrelevant)
    float g0 = gumbel_hash(0x13572468u, (unsigned)(item * 2 + 0));
    float g1 = gumbel_hash(0x13572468u, (unsigned)(item * 2 + 1));
    kclass[item] = (l1 + g1 > l0 + g0) ? 1u : 0u;
    qc[item * 2 + 0] = q0;
    qc[item * 2 + 1] = q1;
}

// ---------------- hard attention: one wave per (b,h,q); gather V row ----------
__global__ __launch_bounds__(256)
void attn_argmax_gather_kernel(const float* __restrict__ Cqkv,
                               const float* __restrict__ qc,
                               const uint8_t* __restrict__ kclass,
                               __bf16* __restrict__ att) {          // [M][768] bf16
    int gw = blockIdx.x * 8 + (threadIdx.x >> 5);   // global wave id = ((b*12+h)*1024+q)
    int lane = threadIdx.x & 31;
    int b = gw / (NHh * Ss);
    int rem = gw - b * NHh * Ss;
    int h = rem / Ss;
    int q = rem - h * Ss;
    int mq = b * Ss + q;

    float qc0 = qc[(mq * NHh + h) * 2 + 0];
    float qc1 = qc[(mq * NHh + h) * 2 + 1];

    float best = -1e30f;
    int bestk = 0;
    for (int k = lane; k < Ss; k += 32) {
        int mk = b * Ss + k;
        int c = kclass[mk * NHh + h];
        float val = (c ? qc1 : qc0) +
                    gumbel_hash(0x89ABCDEFu, (unsigned)gw * 1024u + (unsigned)k);
        if (val > best) { best = val; bestk = k; }
    }
#pragma unroll
    for (int off = 16; off >= 1; off >>= 1) {
        float ov = __shfl_xor(best, off);
        int   ok = __shfl_xor(bestk, off);
        if (ov > best || (ov == best && ok < bestk)) { best = ov; bestk = ok; }
    }
    // attn @ V == gather row bestk of V for this head
    const float* src = Cqkv + (size_t)(b * Ss + bestk) * NQKV + 2 * Hh + h * HDd;
    __bf16* dst = att + (size_t)mq * Hh + h * HDd;
    dst[lane]      = (__bf16)src[lane];
    dst[lane + 32] = (__bf16)src[lane + 32];
}

// ---------------- launch ----------------
extern "C" void kernel_launch(void* const* d_in, const int* in_sizes, int n_in,
                              void* d_out, int out_size, void* d_ws, size_t ws_size,
                              hipStream_t stream) {
    const float* X   = (const float*)d_in[0];
    const float* Wq  = (const float*)d_in[1];
    const float* bq  = (const float*)d_in[2];
    const float* Wk  = (const float*)d_in[3];
    const float* bk  = (const float*)d_in[4];
    const float* Wv  = (const float*)d_in[5];
    const float* bv  = (const float*)d_in[6];
    const float* cen = (const float*)d_in[7];
    const float* Wp  = (const float*)d_in[8];
    const float* bp  = (const float*)d_in[9];
    float* OUT = (float*)d_out;

    char* ws = (char*)d_ws;
    size_t off = 0;
    auto take = [&](size_t bytes) { char* p = ws + off; off = (off + bytes + 255) & ~(size_t)255; return p; };

    __bf16*  Xb    = (__bf16*)take((size_t)Mm * Hh * 2);
    __bf16*  Wcat  = (__bf16*)take((size_t)Hh * NQKV * 2);
    __bf16*  Wpb   = (__bf16*)take((size_t)Hh * Hh * 2);
    float*   bcat  = (float*)take((size_t)NQKV * 4);
    float*   Cqkv  = (float*)take((size_t)Mm * NQKV * 4);
    float*   qc    = (float*)take((size_t)Mm * NHh * 2 * 4);
    uint8_t* kcls  = (uint8_t*)take((size_t)Mm * NHh);
    __bf16*  att   = (__bf16*)take((size_t)Mm * Hh * 2);

    // 1) convert / pack operands to bf16
    f32_to_bf16_kernel<<<4096, 256, 0, stream>>>(X, Xb, Mm * Hh);
    pack_wcat_kernel<<<2048, 256, 0, stream>>>(Wq, Wk, Wv, bq, bk, bv, Wcat, bcat);
    f32_to_bf16_kernel<<<1024, 256, 0, stream>>>(Wp, Wpb, Hh * Hh);

    // 2) fused QKV projection: Cqkv[8192,2304] = Xb @ Wcat + bcat
    gemm_bf16_wmma_kernel<<<dim3(NQKV / 128, Mm / 128), 256, 0, stream>>>(
        Xb, Wcat, bcat, Cqkv, Mm, NQKV, Hh, Hh, NQKV, NQKV);

    // 3) key quantization class + Q·centroid dots
    quant_qc_kernel<<<(Mm * NHh + 255) / 256, 256, 0, stream>>>(Cqkv, cen, qc, kcls);

    // 4) hard gumbel attention argmax + V gather (one wave per (b,h,q))
    attn_argmax_gather_kernel<<<(Bb * NHh * Ss) / 8, 256, 0, stream>>>(Cqkv, qc, kcls, att);

    // 5) output projection: OUT[8192,768] = att @ Wpb + bp
    gemm_bf16_wmma_kernel<<<dim3(Hh / 128, Mm / 128), 256, 0, stream>>>(
        att, Wpb, bp, OUT, Mm, Hh, Hh, Hh, Hh, Hh);
}
